// GroupedQueryAttention_7413113553629
// MI455X (gfx1250) — matrix operand order repaired and checked by reading, not compiled, not run on hardware
//
#include <hip/hip_runtime.h>
#include <hip/hip_bf16.h>

// GQA attention block for MI455X (gfx1250, wave32, WMMA + async/TDM staging).
// Pipeline: cvt->bf16, WMMA GEMM qkv (async A-tile staging), RMSNorm(q,k),
// flash attention (TDM K-tile staging, WMMA), WMMA out-proj.
// Shapes fixed by reference: B=2, S=2048, D=2048, NH=16, NKV=4, DH=128.

typedef __bf16 bf16;
typedef __attribute__((ext_vector_type(16))) __bf16 v16bf;
typedef __attribute__((ext_vector_type(8)))  __bf16 v8bf;
typedef __attribute__((ext_vector_type(4)))  __bf16 v4bf;
typedef __attribute__((ext_vector_type(8)))  float  v8f;
typedef __attribute__((ext_vector_type(4)))  unsigned u32x4;
typedef __attribute__((ext_vector_type(8)))  int      i32x8;
typedef __attribute__((ext_vector_type(4)))  int      i32x4;

union V16U { v16bf v; v8bf h[2]; };

__device__ __forceinline__ bf16 f2bf(float f) {
    // round-to-nearest-even f32 -> bf16
    unsigned u = __float_as_uint(f);
    unsigned r = (u + 0x7FFFu + ((u >> 16) & 1u)) >> 16;
    unsigned short h = (unsigned short)r;
    return __builtin_bit_cast(bf16, h);
}

__device__ __forceinline__ v8f wmma_bf16(v16bf a, v16bf b, v8f c) {
    // D = A(16x32 bf16) x B(32x16 bf16) + C(16x16 f32)
    return __builtin_amdgcn_wmma_f32_16x16x32_bf16(false, a, false, b, (short)0, c, false, false);
}

__device__ __forceinline__ v8f v8f_zero() {
    v8f z;
#pragma unroll
    for (int e = 0; e < 8; ++e) z[e] = 0.0f;
    return z;
}

// --- gfx1250 async global->LDS copy (ASYNCcnt-tracked, no VGPR round trip) ---
__device__ __forceinline__ void async_copy_b128(void* lds, const void* gmem) {
    unsigned l = (unsigned)(uintptr_t)lds;                    // low 32 bits = LDS offset
    unsigned long long g = (unsigned long long)(uintptr_t)gmem;
    asm volatile("global_load_async_to_lds_b128 %0, %1, off" :: "v"(l), "v"(g) : "memory");
}
__device__ __forceinline__ void wait_async0() {
    asm volatile("s_wait_asynccnt 0x0" ::: "memory");
}

// ---------------------------------------------------------------------------
// fp32 -> bf16 elementwise (n divisible by 4)
// ---------------------------------------------------------------------------
__global__ __launch_bounds__(256) void cvt_f32_bf16(const float* __restrict__ s,
                                                    bf16* __restrict__ d, long n4) {
    long idx = (long)blockIdx.x * blockDim.x + threadIdx.x;
    long stride = (long)gridDim.x * blockDim.x;
    for (long i = idx; i < n4; i += stride) {
        float4 x = ((const float4*)s)[i];
        v4bf o;
        o[0] = f2bf(x.x); o[1] = f2bf(x.y); o[2] = f2bf(x.z); o[3] = f2bf(x.w);
        ((v4bf*)d)[i] = o;
    }
}

// ---------------------------------------------------------------------------
// RMSNorm over contiguous 128-wide rows, fp32 in -> bf16 out. One wave per row.
// ---------------------------------------------------------------------------
__global__ __launch_bounds__(256) void rmsnorm128_bf16(const float* __restrict__ src,
                                                       const float* __restrict__ wgt,
                                                       bf16* __restrict__ dst, int rows) {
    const int lane = threadIdx.x & 31;
    const int w    = threadIdx.x >> 5;
    const int row  = blockIdx.x * 8 + w;
    if (row >= rows) return;
    float4 x = ((const float4*)(src + (size_t)row * 128))[lane];
    float ss = x.x * x.x + x.y * x.y + x.z * x.z + x.w * x.w;
#pragma unroll
    for (int m = 1; m < 32; m <<= 1) ss += __shfl_xor(ss, m, 32);
    float inv = rsqrtf(ss * (1.0f / 128.0f) + 1e-6f);
    float4 g = ((const float4*)wgt)[lane];
    v4bf o;
    o[0] = f2bf(x.x * inv * g.x); o[1] = f2bf(x.y * inv * g.y);
    o[2] = f2bf(x.z * inv * g.z); o[3] = f2bf(x.w * inv * g.w);
    ((v4bf*)(dst + (size_t)row * 128))[lane] = o;
}

// ---------------------------------------------------------------------------
// bf16 GEMM with f32 accumulation: C[MxN] = A[MxK] * B[KxN]  (row-major)
// Block tile 128x128, K-stage 64 through LDS. A tile staged with
// global_load_async_to_lds_b128; B tile transposed through VGPRs.
// 8 waves (2x4), each 64x32 (4x2 WMMA 16x16 frags).
// ---------------------------------------------------------------------------
#define BM 128
#define BN 128
#define BKS 64
#define LDA 72
#define LDB 72

__global__ __launch_bounds__(256) void gemm_bf16_f32(const bf16* __restrict__ A,
                                                     const bf16* __restrict__ B,
                                                     float* __restrict__ C,
                                                     int M, int N, int K) {
    __shared__ bf16 As[BM * LDA];   // row-major tile, padded
    __shared__ bf16 Bt[BN * LDB];   // B tile transposed: [n][k]
    const int t    = threadIdx.x;
    const int lane = t & 31;
    const int w    = t >> 5;
    const int m0   = blockIdx.y * BM;
    const int n0   = blockIdx.x * BN;
    const int wm   = (w & 1) * 64;
    const int wn   = (w >> 1) * 32;

    v8f acc[4][2];
#pragma unroll
    for (int i = 0; i < 4; ++i)
#pragma unroll
        for (int j = 0; j < 2; ++j) acc[i][j] = v8f_zero();

    for (int kb = 0; kb < K; kb += BKS) {
        // A tile 128x64 halves: async DMA straight into LDS (no VGPR round trip)
#pragma unroll
        for (int i = 0; i < 4; ++i) {
            int c   = t + 256 * i;
            int row = c >> 3;
            int cc  = (c & 7) << 3;
            async_copy_b128(As + row * LDA + cc,
                            A + (size_t)(m0 + row) * K + kb + cc);
        }
        // B tile 64x128, transposed into LDS (needs VGPR pass for the transpose)
#pragma unroll
        for (int i = 0; i < 4; ++i) {
            int c  = t + 256 * i;
            int kr = c >> 4;
            int cc = (c & 15) << 3;
            const bf16* src = B + (size_t)(kb + kr) * N + n0 + cc;
            __builtin_prefetch(src + (size_t)BKS * N, 0, 1);
            v8bf dchunk = *(const v8bf*)src;
#pragma unroll
            for (int e = 0; e < 8; ++e) Bt[(cc + e) * LDB + kr] = dchunk[e];
        }
        wait_async0();
        __syncthreads();

#pragma unroll
        for (int kk = 0; kk < BKS; kk += 32) {
            v16bf af[4], bfr[2];
#pragma unroll
            for (int i = 0; i < 4; ++i) {
                int row = wm + 16 * i + (lane & 15);
                int ko  = kk + ((lane >> 4) << 3);
                V16U u;
                u.h[0] = *(const v8bf*)(As + row * LDA + ko);
                u.h[1] = *(const v8bf*)(As + row * LDA + ko + 16);
                af[i] = u.v;
            }
#pragma unroll
            for (int j = 0; j < 2; ++j) {
                int col = wn + 16 * j + (lane & 15);
                int ks  = kk + ((lane >> 4) << 3);
                V16U u;
                u.h[0] = *(const v8bf*)(Bt + col * LDB + ks);
                u.h[1] = *(const v8bf*)(Bt + col * LDB + ks + 16);
                bfr[j] = u.v;
            }
#pragma unroll
            for (int i = 0; i < 4; ++i)
#pragma unroll
                for (int j = 0; j < 2; ++j) acc[i][j] = wmma_bf16(af[i], bfr[j], acc[i][j]);
        }
        __syncthreads();
    }

    // epilogue (32-bit indexing: all C buffers < 2^31 elements)
#pragma unroll
    for (int i = 0; i < 4; ++i)
#pragma unroll
        for (int j = 0; j < 2; ++j) {
            unsigned colg  = (unsigned)(n0 + wn + 16 * j + (lane & 15));
            unsigned rbase = (unsigned)(m0 + wm + 16 * i + ((lane >> 4) << 3));
#pragma unroll
            for (int r = 0; r < 8; ++r)
                C[(rbase + r) * (unsigned)N + colg] = acc[i][j][r];
        }
}

// ---------------------------------------------------------------------------
// Flash attention, causal, GQA. One block = (b, head, 128-row q tile).
// 8 waves; wave owns 16 q rows x full 128-key tile (8 N-frags).
// Q frags in registers. K tile DMA'd by the Tensor Data Mover (TDM) with
// hardware padding into the bank-conflict-free LDS layout; V^T + per-wave P
// in dynamic LDS. Both Q@K^T and P@V via v_wmma_f32_16x16x32_bf16.
// ---------------------------------------------------------------------------
#define SLD 136
#define PLD 136
#define NEG_BIG (-3.0e38f)

__global__ __launch_bounds__(256) void attn_flash(const bf16* __restrict__ qb,
                                                  const bf16* __restrict__ kb,
                                                  const bf16* __restrict__ vb,
                                                  bf16* __restrict__ aob,
                                                  int S) {
    extern __shared__ bf16 smem[];
    bf16* Ks = smem;                 // [128][SLD]  key-major (dims contiguous)
    bf16* Vt = Ks + 128 * SLD;       // [128][SLD]  dim-major (keys contiguous)
    bf16* Ps = Vt + 128 * SLD;       // 8 waves x [16][PLD]

    const int t    = threadIdx.x;
    const int lane = t & 31;
    const int w    = t >> 5;
    const int qt   = blockIdx.x;
    const int head = blockIdx.y;
    const int b    = blockIdx.z;
    const int kv   = head >> 2;          // groups = 4
    const int q0   = qt * 128;
    const int rowbase = q0 + 16 * w;
    const int Dq = 16 * 128;             // 2048
    const int Dk = 4 * 128;              // 512
    const float scale = 0.0883883476483184f; // 1/sqrt(128)

    // Q fragments (16 rows x 128 dims = 4 A-frags), kept in registers
    v16bf qf[4];
    {
        int row = rowbase + (lane & 15);
        const bf16* qp = qb + (size_t)(b * S + row) * Dq + head * 128;
        int ko = (lane >> 4) << 3;
#pragma unroll
        for (int f = 0; f < 4; ++f) {
            V16U u;
            u.h[0] = *(const v8bf*)(qp + 32 * f + ko);
            u.h[1] = *(const v8bf*)(qp + 32 * f + 16 + ko);
            qf[f] = u.v;
        }
    }

    float mrun[8], lrun[8];
    v8f Oa[8];
#pragma unroll
    for (int r = 0; r < 8; ++r) { mrun[r] = NEG_BIG; lrun[r] = 0.0f; Oa[r] = v8f_zero(); }

    bf16* Pw = Ps + w * 16 * PLD;

    for (int kt = 0; kt <= qt; ++kt) {
        const int k0 = kt * 128;

        // ---- K tile via Tensor Data Mover: 2D tile 128(dims) x 128(keys),
        //      data_size=2B, row stride Dk=512 elems, LDS padding 4 DWORDs
        //      every 64 DWORDs (-> SLD=136 halves row pitch). Issued by wave 0.
        if (w == 0) {
            const bf16* gK = kb + (size_t)(b * S + k0) * Dk + kv * 128;
            unsigned long long ga = (unsigned long long)(uintptr_t)gK;
            u32x4 g0;
            g0[0] = 1u;                                        // count=1 (valid user D#)
            g0[1] = (unsigned)(uintptr_t)Ks;                   // lds_addr
            g0[2] = (unsigned)ga;                              // global_addr[31:0]
            g0[3] = (unsigned)((ga >> 32) & 0x1FFFFFFu) | 0x80000000u; // addr[56:32] | type=2
            i32x8 g1;
            g1[0] = (1 << 16)        // data_size = 2 bytes
                  | (1 << 20)        // pad_enable
                  | (5 << 22)        // pad_interval: 64 DWORDs
                  | (3 << 25);       // pad_amount: 4 DWORDs
            g1[1] = 128 << 16;       // tensor_dim0[15:0] = 128
            g1[2] = 128 << 16;       // tensor_dim1[15:0] = 128
            g1[3] = 128 << 16;       // tile_dim0 = 128
            g1[4] = 128;             // tile_dim1 = 128, tile_dim2 = 0
            g1[5] = Dk;              // tensor_dim0_stride = 512
            g1[6] = 0;
            g1[7] = 0;
            i32x4 gz4;
            gz4[0] = 0; gz4[1] = 0; gz4[2] = 0; gz4[3] = 0;
            i32x8 gz8;
#pragma unroll
            for (int e = 0; e < 8; ++e) gz8[e] = 0;
            // 6-arg toolchain variant: (g0, g1, g2, g3, g4, cpol)
            __builtin_amdgcn_tensor_load_to_lds(g0, g1, gz4, gz4, gz8, 0);
        }

        // ---- V tile transposed into LDS (transpose needs the VGPR pass)
#pragma unroll
        for (int i = 0; i < 8; ++i) {
            int c   = t + 256 * i;          // 0..2047
            int key = c >> 4;
            int dc  = (c & 15) << 3;
            v8bf vv = *(const v8bf*)(vb + (size_t)(b * S + k0 + key) * Dk + kv * 128 + dc);
#pragma unroll
            for (int e = 0; e < 8; ++e) Vt[(dc + e) * SLD + key] = vv[e];
        }
        if (w == 0) __builtin_amdgcn_s_wait_tensorcnt(0);
        __syncthreads();

        // scores S = Q K^T  (16 x 128 per wave)
        v8f sc[8];
#pragma unroll
        for (int j = 0; j < 8; ++j) sc[j] = v8f_zero();
#pragma unroll
        for (int ks = 0; ks < 4; ++ks) {
            v16bf a = qf[ks];
#pragma unroll
            for (int j = 0; j < 8; ++j) {
                int key = 16 * j + (lane & 15);
                int kd  = 32 * ks + ((lane >> 4) << 3);
                V16U u;
                u.h[0] = *(const v8bf*)(Ks + key * SLD + kd);
                u.h[1] = *(const v8bf*)(Ks + key * SLD + kd + 16);
                sc[j] = wmma_bf16(a, u.v, sc[j]);
            }
        }

        // scale + causal mask (diagonal tile only)
        const bool diag = (kt == qt);
#pragma unroll
        for (int j = 0; j < 8; ++j)
#pragma unroll
            for (int r = 0; r < 8; ++r) {
                float s = sc[j][r] * scale;
                if (diag) {
                    int col = k0 + 16 * j + (lane & 15);
                    int row = rowbase + ((lane >> 4) << 3) + r;
                    if (col > row) s = NEG_BIG;
                }
                sc[j][r] = s;
            }

        // online softmax per row (half-wave reductions), P -> LDS (A-layout staging)
#pragma unroll
        for (int r = 0; r < 8; ++r) {
            float v = sc[0][r];
#pragma unroll
            for (int j = 1; j < 8; ++j) v = fmaxf(v, sc[j][r]);
#pragma unroll
            for (int m = 1; m < 16; m <<= 1) v = fmaxf(v, __shfl_xor(v, m, 32));
            float mnew  = fmaxf(mrun[r], v);
            float alpha = __expf(mrun[r] - mnew);
            float lsum  = 0.0f;
            int mrow = ((lane >> 4) << 3) + r;
#pragma unroll
            for (int j = 0; j < 8; ++j) {
                float p = __expf(sc[j][r] - mnew);
                lsum += p;
                Pw[mrow * PLD + 16 * j + (lane & 15)] = f2bf(p);
            }
#pragma unroll
            for (int m = 1; m < 16; m <<= 1) lsum += __shfl_xor(lsum, m, 32);
            lrun[r] = lrun[r] * alpha + lsum;
            mrun[r] = mnew;
#pragma unroll
            for (int j = 0; j < 8; ++j) Oa[j][r] *= alpha;
        }

        // O += P @ V   (P: 16x128 keys, V^T tiles give B-layout over dims)
#pragma unroll
        for (int ks = 0; ks < 4; ++ks) {
            V16U ua;
            int prow = lane & 15;
            int ko   = (lane >> 4) << 3;
            ua.h[0] = *(const v8bf*)(Pw + prow * PLD + 32 * ks + ko);
            ua.h[1] = *(const v8bf*)(Pw + prow * PLD + 32 * ks + 16 + ko);
            v16bf a = ua.v;
#pragma unroll
            for (int j = 0; j < 8; ++j) {
                int dim = 16 * j + (lane & 15);
                int kst = 32 * ks + ((lane >> 4) << 3);
                V16U ub;
                ub.h[0] = *(const v8bf*)(Vt + dim * SLD + kst);
                ub.h[1] = *(const v8bf*)(Vt + dim * SLD + kst + 16);
                Oa[j] = wmma_bf16(a, ub.v, Oa[j]);
            }
        }
        __syncthreads();   // before next tile overwrites Ks/Vt
    }

    // normalize and write bf16 attention output (B,S,D) with col = head*128 + d
#pragma unroll
    for (int r = 0; r < 8; ++r) {
        float inv = 1.0f / lrun[r];
        unsigned row = (unsigned)(rowbase + ((lane >> 4) << 3) + r);
        bf16* op = aob + (unsigned)(b * S) * (unsigned)Dq + row * (unsigned)Dq + head * 128;
#pragma unroll
        for (int j = 0; j < 8; ++j) op[16 * j + (lane & 15)] = f2bf(Oa[j][r] * inv);
    }
}

// ---------------------------------------------------------------------------
// host-side launcher
// ---------------------------------------------------------------------------
extern "C" void kernel_launch(void* const* d_in, const int* in_sizes, int n_in,
                              void* d_out, int out_size, void* d_ws, size_t ws_size,
                              hipStream_t stream) {
    (void)in_sizes; (void)n_in; (void)out_size; (void)ws_size;
    const float* x   = (const float*)d_in[0];
    const float* wq  = (const float*)d_in[1];
    const float* wk  = (const float*)d_in[2];
    const float* wv  = (const float*)d_in[3];
    const float* wo  = (const float*)d_in[4];
    const float* qnw = (const float*)d_in[5];
    const float* knw = (const float*)d_in[6];
    float* out = (float*)d_out;

    const int B = 2, S = 2048, D = 2048, NH = 16, NKV = 4;
    const size_t M  = (size_t)B * S;     // 4096 rows
    const size_t Dk = (size_t)NKV * 128; // 512

    // workspace carve-out (256B aligned); total ~130 MB
    char* ws = (char*)d_ws;
    size_t off = 0;
    auto carve = [&](size_t bytes) -> void* {
        void* p = ws + off;
        off += (bytes + 255) & ~(size_t)255;
        return p;
    };
    bf16*  xb  = (bf16*)carve(M * D * 2);
    bf16*  wqb = (bf16*)carve((size_t)D * D * 2);
    bf16*  wkb = (bf16*)carve((size_t)D * Dk * 2);
    bf16*  wvb = (bf16*)carve((size_t)D * Dk * 2);
    bf16*  wob = (bf16*)carve((size_t)D * D * 2);
    float* qf  = (float*)carve(M * D * 4);
    float* kf  = (float*)carve(M * Dk * 4);
    float* vf  = (float*)carve(M * Dk * 4);
    bf16*  qb2 = (bf16*)carve(M * D * 2);
    bf16*  kb2 = (bf16*)carve(M * Dk * 2);
    bf16*  vb2 = (bf16*)carve(M * Dk * 2);
    bf16*  aob = (bf16*)carve(M * D * 2);

    // 1) downconvert activations + weights to bf16
    cvt_f32_bf16<<<2048, 256, 0, stream>>>(x,  xb,  (long)(M * D / 4));
    cvt_f32_bf16<<<2048, 256, 0, stream>>>(wq, wqb, (long)((size_t)D * D / 4));
    cvt_f32_bf16<<<1024, 256, 0, stream>>>(wk, wkb, (long)((size_t)D * Dk / 4));
    cvt_f32_bf16<<<1024, 256, 0, stream>>>(wv, wvb, (long)((size_t)D * Dk / 4));
    cvt_f32_bf16<<<2048, 256, 0, stream>>>(wo, wob, (long)((size_t)D * D / 4));

    // 2) QKV projections (WMMA GEMM, f32 accum, async A-tile staging)
    gemm_bf16_f32<<<dim3(D / BN, M / BM), 256, 0, stream>>>(xb, wqb, qf, (int)M, D, D);
    gemm_bf16_f32<<<dim3((int)Dk / BN, M / BM), 256, 0, stream>>>(xb, wkb, kf, (int)M, (int)Dk, D);
    gemm_bf16_f32<<<dim3((int)Dk / BN, M / BM), 256, 0, stream>>>(xb, wvb, vf, (int)M, (int)Dk, D);

    // 3) qk RMSNorm (over d_head=128) + bf16; V straight to bf16
    rmsnorm128_bf16<<<(int)(M * NH / 8), 256, 0, stream>>>(qf, qnw, qb2, (int)(M * NH));
    rmsnorm128_bf16<<<(int)(M * NKV / 8), 256, 0, stream>>>(kf, knw, kb2, (int)(M * NKV));
    cvt_f32_bf16<<<1024, 256, 0, stream>>>(vf, vb2, (long)(M * Dk / 4));

    // 4) causal GQA flash attention (WMMA + TDM), dynamic LDS 104448 B
    size_t shmem = (size_t)(2 * 128 * SLD + 8 * 16 * PLD) * sizeof(bf16);
    attn_flash<<<dim3(S / 128, NH, B), 256, shmem, stream>>>(qb2, kb2, vb2, aob, S);

    // 5) output projection -> f32 result
    gemm_bf16_f32<<<dim3(D / BN, M / BM), 256, 0, stream>>>(aob, wob, out, (int)M, D, D);
}
